// GNN_45835890983353
// MI455X (gfx1250) — compile-verified, matrix operand order checked by reading, not run on hardware
//
#include <hip/hip_runtime.h>
#include <cstdint>

typedef __attribute__((ext_vector_type(16))) __bf16 v16bf;
typedef __attribute__((ext_vector_type(8)))  float  v8f;

#define DDIM 128

__device__ __forceinline__ unsigned short f32_to_bf16(float f) {
  unsigned int u = __float_as_uint(f);
  u += 0x7FFFu + ((u >> 16) & 1u);   // round-to-nearest-even
  return (unsigned short)(u >> 16);
}
__device__ __forceinline__ float bf16_to_f32(unsigned short h) {
  return __uint_as_float(((unsigned int)h) << 16);
}

// ---------------- utility kernels ----------------
__global__ void k_zero_f32v4(float4* __restrict__ p, int n4) {
  int i = blockIdx.x * blockDim.x + threadIdx.x;
  if (i < n4) p[i] = make_float4(0.f, 0.f, 0.f, 0.f);
}

__global__ void k_degree(const int* __restrict__ dst, float* __restrict__ deg, int E) {
  int e = blockIdx.x * blockDim.x + threadIdx.x;
  if (e < E) unsafeAtomicAdd(&deg[dst[e]], 1.0f);
}

__global__ void k_invdeg(float* __restrict__ deg, int n) {
  int i = blockIdx.x * blockDim.x + threadIdx.x;
  if (i < n) deg[i] = 1.0f / fmaxf(deg[i], 1.0f);
}

__global__ void k_f32_to_bf16(const float* __restrict__ src,
                              unsigned short* __restrict__ dstp, int n) {
  int i = blockIdx.x * blockDim.x + threadIdx.x;
  if (i < n) dstp[i] = f32_to_bf16(src[i]);
}

// ---------------- h = relu(x @ Wp^T + bp), bf16 in/out ----------------
// Block = 256 threads = 8 waves; block handles 16 rows x 128 cols;
// wave w owns column tile [16w, 16w+16).
__global__ __launch_bounds__(256)
void k_project(const unsigned short* __restrict__ xb,   // N x 128 bf16
               const unsigned short* __restrict__ Wb,   // 128 x 128 bf16 (row n, col k)
               const float* __restrict__ bias,          // 128
               unsigned short* __restrict__ hb,         // N x 128 bf16
               int nrows) {
  __shared__ __align__(32) unsigned short As[16 * DDIM];
  const int tid  = threadIdx.x;
  const int wave = tid >> 5;
  const int lane = tid & 31;
  const int rbase = blockIdx.x * 16;
  const bool full = (rbase + 16 <= nrows);   // block-uniform
  {  // stage 16x128 bf16 A tile into LDS (16B per thread)
    int r  = tid >> 4;
    int cb = (tid & 15) * 8;
    int gr = rbase + r; if (gr >= nrows) gr = nrows - 1;
    *reinterpret_cast<uint4*>(&As[r * DDIM + cb]) =
        *reinterpret_cast<const uint4*>(&xb[(size_t)gr * DDIM + cb]);
  }
  __syncthreads();
  const int half = lane >> 4;    // K-range selector
  const int l16  = lane & 15;    // row (A) / column (B)
  v8f c = {0.f, 0.f, 0.f, 0.f, 0.f, 0.f, 0.f, 0.f};
#pragma unroll
  for (int kb = 0; kb < 4; ++kb) {
    const int kOff = kb * 32 + half * 16;
    v16bf a = *reinterpret_cast<const v16bf*>(&As[l16 * DDIM + kOff]);
    v16bf b = *reinterpret_cast<const v16bf*>(&Wb[(size_t)(wave * 16 + l16) * DDIM + kOff]);
    c = __builtin_amdgcn_wmma_f32_16x16x32_bf16(false, a, false, b, (short)0, c, false, false);
  }
  const int col = wave * 16 + l16;
  const float bv = bias[col];
  unsigned short* hp = &hb[(size_t)(rbase + half * 8) * DDIM + col];
  if (full) {
#pragma unroll
    for (int v = 0; v < 8; ++v)
      hp[(size_t)v * DDIM] = f32_to_bf16(fmaxf(c[v] + bv, 0.0f));
  } else {
#pragma unroll
    for (int v = 0; v < 8; ++v)
      if (rbase + v + half * 8 < nrows)
        hp[(size_t)v * DDIM] = f32_to_bf16(fmaxf(c[v] + bv, 0.0f));
  }
}

// ---------------- agg[dst] += h[src] (one wave per edge) ----------------
__global__ __launch_bounds__(256)
void k_scatter(const unsigned short* __restrict__ hb,
               const int* __restrict__ src,
               const int* __restrict__ dst,
               float* __restrict__ agg, int E) {
  const int w = blockIdx.x * 8 + (threadIdx.x >> 5);
  if (w >= E) return;
  const int lane = threadIdx.x & 31;
  const int s = src[w];
  const int t = dst[w];
  const ushort4 hv = *reinterpret_cast<const ushort4*>(&hb[(size_t)s * DDIM + lane * 4]);
  float* a = &agg[(size_t)t * DDIM + lane * 4];
  unsafeAtomicAdd(a + 0, bf16_to_f32(hv.x));
  unsafeAtomicAdd(a + 1, bf16_to_f32(hv.y));
  unsafeAtomicAdd(a + 2, bf16_to_f32(hv.z));
  unsafeAtomicAdd(a + 3, bf16_to_f32(hv.w));
}

// ------- out = (agg*inv_deg) @ Wl^T + bl + x @ Wr^T; [normalize+relu] -------
template <bool FINAL>
__global__ __launch_bounds__(256)
void k_combine(const float* __restrict__ agg,
               const float* __restrict__ inv_deg,
               const unsigned short* __restrict__ xb,
               const unsigned short* __restrict__ Wlb,
               const float* __restrict__ bias,
               const unsigned short* __restrict__ Wrb,
               unsigned short* xb_out,      // !FINAL: next-layer x (bf16)
               float* out_f32,              // FINAL: fp32 output
               int nrows) {
  __shared__ __align__(32) unsigned short Aagg[16 * DDIM];
  __shared__ __align__(32) unsigned short Ax[16 * DDIM];
  __shared__ float norms[16];
  const int tid  = threadIdx.x;
  const int wave = tid >> 5;
  const int lane = tid & 31;
  const int rbase = blockIdx.x * 16;
  const bool full = (rbase + 16 <= nrows);   // block-uniform
  {
    int r  = tid >> 4;
    int cb = (tid & 15) * 8;
    int gr = rbase + r; if (gr >= nrows) gr = nrows - 1;
    const float invd = inv_deg[gr];
    float4 v0 = *reinterpret_cast<const float4*>(&agg[(size_t)gr * DDIM + cb]);
    float4 v1 = *reinterpret_cast<const float4*>(&agg[(size_t)gr * DDIM + cb + 4]);
    unsigned short* s = &Aagg[r * DDIM + cb];
    s[0] = f32_to_bf16(v0.x * invd); s[1] = f32_to_bf16(v0.y * invd);
    s[2] = f32_to_bf16(v0.z * invd); s[3] = f32_to_bf16(v0.w * invd);
    s[4] = f32_to_bf16(v1.x * invd); s[5] = f32_to_bf16(v1.y * invd);
    s[6] = f32_to_bf16(v1.z * invd); s[7] = f32_to_bf16(v1.w * invd);
    *reinterpret_cast<uint4*>(&Ax[r * DDIM + cb]) =
        *reinterpret_cast<const uint4*>(&xb[(size_t)gr * DDIM + cb]);
  }
  if (!FINAL && tid < 16) norms[tid] = 0.0f;
  __syncthreads();
  const int half = lane >> 4;
  const int l16  = lane & 15;
  v8f c = {0.f, 0.f, 0.f, 0.f, 0.f, 0.f, 0.f, 0.f};
#pragma unroll
  for (int kb = 0; kb < 4; ++kb) {
    const int kOff = kb * 32 + half * 16;
    v16bf aA = *reinterpret_cast<const v16bf*>(&Aagg[l16 * DDIM + kOff]);
    v16bf bL = *reinterpret_cast<const v16bf*>(&Wlb[(size_t)(wave * 16 + l16) * DDIM + kOff]);
    c = __builtin_amdgcn_wmma_f32_16x16x32_bf16(false, aA, false, bL, (short)0, c, false, false);
    v16bf aX = *reinterpret_cast<const v16bf*>(&Ax[l16 * DDIM + kOff]);
    v16bf bR = *reinterpret_cast<const v16bf*>(&Wrb[(size_t)(wave * 16 + l16) * DDIM + kOff]);
    c = __builtin_amdgcn_wmma_f32_16x16x32_bf16(false, aX, false, bR, (short)0, c, false, false);
  }
  const int col = wave * 16 + l16;
  const float bv = bias[col];
  float o[8];
#pragma unroll
  for (int v = 0; v < 8; ++v) o[v] = c[v] + bv;

  if (FINAL) {
    float* op = &out_f32[(size_t)(rbase + half * 8) * DDIM + col];
    if (full) {
#pragma unroll
      for (int v = 0; v < 8; ++v) op[(size_t)v * DDIM] = o[v];
    } else {
#pragma unroll
      for (int v = 0; v < 8; ++v)
        if (rbase + v + half * 8 < nrows) op[(size_t)v * DDIM] = o[v];
    }
  } else {
    // per-row sum of squares: butterfly across the 16-lane column group,
    // then one LDS atomic per (wave, half, v) -> 128 ds_add per block
#pragma unroll
    for (int v = 0; v < 8; ++v) {
      float s2 = o[v] * o[v];
      s2 += __shfl_xor(s2, 1, 16);
      s2 += __shfl_xor(s2, 2, 16);
      s2 += __shfl_xor(s2, 4, 16);
      s2 += __shfl_xor(s2, 8, 16);
      if (l16 == 0) atomicAdd(&norms[v + half * 8], s2);
    }
    __syncthreads();
    unsigned short* xp = &xb_out[(size_t)(rbase + half * 8) * DDIM + col];
    if (full) {
#pragma unroll
      for (int v = 0; v < 8; ++v) {
        float nrm = fmaxf(sqrtf(norms[v + half * 8]), 1e-12f);
        xp[(size_t)v * DDIM] = f32_to_bf16(fmaxf(o[v] / nrm, 0.0f));
      }
    } else {
#pragma unroll
      for (int v = 0; v < 8; ++v) {
        if (rbase + v + half * 8 < nrows) {
          float nrm = fmaxf(sqrtf(norms[v + half * 8]), 1e-12f);
          xp[(size_t)v * DDIM] = f32_to_bf16(fmaxf(o[v] / nrm, 0.0f));
        }
      }
    }
  }
}

// ---------------- host launcher ----------------
extern "C" void kernel_launch(void* const* d_in, const int* in_sizes, int n_in,
                              void* d_out, int out_size, void* d_ws, size_t ws_size,
                              hipStream_t stream) {
  const float* x    = (const float*)d_in[0];
  const int*   edge = (const int*)  d_in[1];
  const float* Wp   = (const float*)d_in[2];
  const float* bp   = (const float*)d_in[3];
  const float* Wl   = (const float*)d_in[4];
  const float* bl   = (const float*)d_in[5];
  const float* Wr   = (const float*)d_in[6];
  const float* WlO  = (const float*)d_in[7];
  const float* blO  = (const float*)d_in[8];
  const float* WrO  = (const float*)d_in[9];
  float* out = (float*)d_out;

  const int D = DDIM;
  const int N = in_sizes[0] / D;
  const int E = in_sizes[1] / 2;
  const int* src = edge;
  const int* dst = edge + E;

  // workspace carve-up (~103 MB)
  char* w = (char*)d_ws;
  float* invdeg = (float*)w;                 w += (size_t)N * sizeof(float);
  w = (char*)(((uintptr_t)w + 255) & ~(uintptr_t)255);
  unsigned short* wts  = (unsigned short*)w; w += (size_t)8 * D * D * sizeof(unsigned short);
  unsigned short* bufA = (unsigned short*)w; w += (size_t)N * D * sizeof(unsigned short);
  unsigned short* bufB = (unsigned short*)w; w += (size_t)N * D * sizeof(unsigned short);
  float* agg = (float*)w;

  unsigned short* Wp_b  = wts;
  unsigned short* Wl_b  = wts + 2 * D * D;
  unsigned short* Wr_b  = wts + 4 * D * D;
  unsigned short* WlO_b = wts + 6 * D * D;
  unsigned short* WrO_b = wts + 7 * D * D;

  const int TB = 256;
  auto cdiv = [](int a, int b) { return (a + b - 1) / b; };

  // degrees -> inverse degrees (deg buffer zeroed every call: determinism)
  k_zero_f32v4<<<cdiv(N / 4, TB), TB, 0, stream>>>((float4*)invdeg, N / 4);
  k_degree<<<cdiv(E, TB), TB, 0, stream>>>(dst, invdeg, E);
  k_invdeg<<<cdiv(N, TB), TB, 0, stream>>>(invdeg, N);

  // bf16 copies of weights and x
  k_f32_to_bf16<<<cdiv(2 * D * D, TB), TB, 0, stream>>>(Wp,  Wp_b,  2 * D * D);
  k_f32_to_bf16<<<cdiv(2 * D * D, TB), TB, 0, stream>>>(Wl,  Wl_b,  2 * D * D);
  k_f32_to_bf16<<<cdiv(2 * D * D, TB), TB, 0, stream>>>(Wr,  Wr_b,  2 * D * D);
  k_f32_to_bf16<<<cdiv(D * D, TB), TB, 0, stream>>>(WlO, WlO_b, D * D);
  k_f32_to_bf16<<<cdiv(D * D, TB), TB, 0, stream>>>(WrO, WrO_b, D * D);
  k_f32_to_bf16<<<cdiv(N * D, TB), TB, 0, stream>>>(x, bufA, N * D);

  const int rowBlocks  = cdiv(N, 16);
  const int edgeBlocks = cdiv(E, 8);
  const int nd4 = (N * D) / 4;

  unsigned short* cur = bufA;  // current-layer x (bf16)
  unsigned short* oth = bufB;  // h scratch, then next-layer x (h dead by then)
  for (int i = 0; i < 2; ++i) {
    k_project<<<rowBlocks, TB, 0, stream>>>(cur, Wp_b + (size_t)i * D * D, bp + i * D, oth, N);
    k_zero_f32v4<<<cdiv(nd4, TB), TB, 0, stream>>>((float4*)agg, nd4);
    k_scatter<<<edgeBlocks, TB, 0, stream>>>(oth, src, dst, agg, E);
    k_combine<false><<<rowBlocks, TB, 0, stream>>>(
        agg, invdeg, cur, Wl_b + (size_t)i * D * D, bl + i * D,
        Wr_b + (size_t)i * D * D, oth, nullptr, N);
    unsigned short* t = cur; cur = oth; oth = t;
  }
  // final layer: message = x itself (no project), fp32 out, no normalize
  k_zero_f32v4<<<cdiv(nd4, TB), TB, 0, stream>>>((float4*)agg, nd4);
  k_scatter<<<edgeBlocks, TB, 0, stream>>>(cur, src, dst, agg, E);
  k_combine<true><<<rowBlocks, TB, 0, stream>>>(
      agg, invdeg, cur, WlO_b, blO, WrO_b, nullptr, out, N);

  (void)n_in; (void)out_size; (void)ws_size;
}
